// Lex2d_61108794687739
// MI455X (gfx1250) — matrix operand order repaired and checked by reading, not compiled, not run on hardware
//
#include <hip/hip_runtime.h>
#include <hip/hip_bf16.h>
#include <math.h>

#define D 128
#define WSTR 136   // padded bf16 row stride for weight tiles in LDS (bank de-conflict)
#define XSTR 136   // padded bf16 row stride for activation tiles in LDS
#define MSTR 130   // padded f32 row stride for the LN staging tile

typedef __attribute__((ext_vector_type(16))) __bf16 v16bf;
typedef __attribute__((ext_vector_type(8)))  __bf16 v8bf;
typedef __attribute__((ext_vector_type(8)))  float  v8f;

// ---------- scalar helpers ----------
static __device__ __forceinline__ __bf16 f2bf(float f) {
  unsigned u = __builtin_bit_cast(unsigned, f);
  u += 0x7FFFu + ((u >> 16) & 1u);            // round-to-nearest-even
  return __builtin_bit_cast(__bf16, (unsigned short)(u >> 16));
}
static __device__ __forceinline__ float eluf(float x)   { return x > 0.f ? x : expm1f(x); }
static __device__ __forceinline__ float leakyf(float x) { return x >= 0.f ? x : 0.01f * x; }

static __device__ __forceinline__ v8f wmma_bf16(v16bf a, v16bf b, v8f c) {
  return __builtin_amdgcn_wmma_f32_16x16x32_bf16(false, a, false, b, (short)0, c, false, false);
}

// A fragment: 16x32 bf16 tile. Per ISA layout: lanes 0-15 rows 0-15 / K-base 0,
// lanes 16-31 same rows / K-base +8; halves 0..7 -> K, halves 8..15 -> K+16.
static __device__ __forceinline__ v16bf load_afrag(const __bf16* sA, int waveRow, int c, int lane) {
  int m  = waveRow + (lane & 15);
  int kb = c * 32 + ((lane >> 4) << 3);
  const __bf16* p = sA + m * XSTR + kb;
  v8bf lo = *(const v8bf*)p;
  v8bf hi = *(const v8bf*)(p + 16);
  return __builtin_shufflevector(lo, hi, 0,1,2,3,4,5,6,7,8,9,10,11,12,13,14,15);
}

// B fragment: 32x16 tile of W^T (B[k][n] = W[n][k]). Lanes 0-15 cols 0-15 / K-base 0,
// lanes 16-31 same cols / K-base +8; halves 0..7 -> K, halves 8..15 -> K+16 (matches A).
static __device__ __forceinline__ v16bf load_bfrag(const __bf16* sW, int ct, int c, int lane) {
  int n  = ct * 16 + (lane & 15);
  int kb = c * 32 + ((lane >> 4) << 3);
  const __bf16* p = sW + n * WSTR + kb;
  v8bf lo = *(const v8bf*)p;
  v8bf hi = *(const v8bf*)(p + 16);
  return __builtin_shufflevector(lo, hi, 0,1,2,3,4,5,6,7,8,9,10,11,12,13,14,15);
}

// ---------- LDS staging ----------
static __device__ __forceinline__ void stage_w(__bf16* sW, const float* __restrict__ W, int tid) {
  for (int i = tid; i < (D * D) / 4; i += 256) {
    float4 v = ((const float4*)W)[i];
    int e = i * 4, row = e >> 7, col = e & 127;
    __bf16* p = sW + row * WSTR + col;
    p[0] = f2bf(v.x); p[1] = f2bf(v.y); p[2] = f2bf(v.z); p[3] = f2bf(v.w);
  }
}

static __device__ __forceinline__ void stage_x(__bf16* sX, const float* __restrict__ A,
                                               const float* __restrict__ A2,
                                               long long rowBase, int M, int tid) {
  for (int i = tid; i < (D * D) / 4; i += 256) {
    int r = i >> 5;                       // 32 float4 per 128-wide row
    long long row = rowBase + r;
    float4 v = make_float4(0.f, 0.f, 0.f, 0.f);
    if (row < M) {
      v = ((const float4*)A)[(size_t)row * 32 + (i & 31)];
      if (A2) {
        float4 w = ((const float4*)A2)[(size_t)row * 32 + (i & 31)];
        v.x += w.x; v.y += w.y; v.z += w.z; v.w += w.w;
      }
    }
    __bf16* p = sX + r * XSTR + ((i & 31) << 2);
    p[0] = f2bf(v.x); p[1] = f2bf(v.y); p[2] = f2bf(v.z); p[3] = f2bf(v.w);
  }
}

// Per-wave 16x128 output tile: 8 column tiles x 4 K-chunks = 32 WMMAs.
static __device__ __forceinline__ void gemm_tile(const __bf16* sX, const __bf16* sW,
                                                 int wave, int lane, v8f acc[8]) {
  const v8f vz = {0.f,0.f,0.f,0.f,0.f,0.f,0.f,0.f};
  #pragma unroll
  for (int ct = 0; ct < 8; ++ct) acc[ct] = vz;
  #pragma unroll
  for (int c = 0; c < 4; ++c) {
    v16bf a = load_afrag(sX, wave * 16, c, lane);
    #pragma unroll
    for (int ct = 0; ct < 8; ++ct) {
      v16bf b = load_bfrag(sW, ct, c, lane);
      acc[ct] = wmma_bf16(a, b, acc[ct]);
    }
  }
}

static __device__ __forceinline__ int lowerb(const int* __restrict__ b, int n, int v) {
  int lo = 0, hi = n;
  while (lo < hi) { int mid = (lo + hi) >> 1; if (b[mid] < v) lo = mid + 1; else hi = mid; }
  return lo;
}

// ---------- fused transfer: out = elu( LN(elu(X@W1^T)) @ W2^T + X ) ----------
__global__ __launch_bounds__(256) void k_transfer(const float* __restrict__ X,
    const float* __restrict__ W1, const float* __restrict__ W2,
    float* __restrict__ out, int M) {
  __shared__ __bf16 sW1[D * WSTR];
  __shared__ __bf16 sW2[D * WSTR];
  __shared__ __bf16 sX [D * XSTR];
  __shared__ float  sMid[D * MSTR];
  int tid = threadIdx.x, wave = tid >> 5, lane = tid & 31;
  long long rowBase = (long long)blockIdx.x * 128;
  stage_w(sW1, W1, tid);
  stage_w(sW2, W2, tid);
  stage_x(sX, X, nullptr, rowBase, M, tid);
  __syncthreads();

  v8f acc[8];
  gemm_tile(sX, sW1, wave, lane, acc);
  int mloc = (lane >> 4) << 3;
  #pragma unroll
  for (int ct = 0; ct < 8; ++ct) {
    int n = ct * 16 + (lane & 15);
    #pragma unroll
    for (int r = 0; r < 8; ++r)
      sMid[(wave * 16 + mloc + r) * MSTR + n] = eluf(acc[ct][r]);
  }
  __syncthreads();

  if (tid < 128) {                                    // LayerNorm per row
    float mean = 0.f;
    #pragma unroll 8
    for (int j = 0; j < D; ++j) mean += sMid[tid * MSTR + j];
    mean *= (1.f / D);
    float var = 0.f;
    #pragma unroll 8
    for (int j = 0; j < D; ++j) { float d = sMid[tid * MSTR + j] - mean; var += d * d; }
    var *= (1.f / D);
    float rs = __frsqrt_rn(var + 1e-5f);
    #pragma unroll 8
    for (int j = 0; j < D; ++j) sX[tid * XSTR + j] = f2bf((sMid[tid * MSTR + j] - mean) * rs);
  }
  __syncthreads();

  gemm_tile(sX, sW2, wave, lane, acc);
  #pragma unroll
  for (int ct = 0; ct < 8; ++ct) {
    int n = ct * 16 + (lane & 15);
    #pragma unroll
    for (int r = 0; r < 8; ++r) {
      long long gr = rowBase + wave * 16 + mloc + r;
      if (gr < M) out[(size_t)gr * D + n] = eluf(acc[ct][r] + X[(size_t)gr * D + n]);
    }
  }
}

// ---------- generic WMMA GEMM: out = (A [+ A2]) @ W^T ----------
__global__ __launch_bounds__(256) void k_gemm(const float* __restrict__ A,
    const float* __restrict__ A2, const float* __restrict__ W,
    float* __restrict__ out, int M) {
  __shared__ __bf16 sW[D * WSTR];
  __shared__ __bf16 sX[D * XSTR];
  int tid = threadIdx.x, wave = tid >> 5, lane = tid & 31;
  long long rowBase = (long long)blockIdx.x * 128;
  stage_w(sW, W, tid);
  stage_x(sX, A, A2, rowBase, M, tid);
  __syncthreads();
  v8f acc[8];
  gemm_tile(sX, sW, wave, lane, acc);
  int mloc = (lane >> 4) << 3;
  #pragma unroll
  for (int ct = 0; ct < 8; ++ct) {
    int n = ct * 16 + (lane & 15);
    #pragma unroll
    for (int r = 0; r < 8; ++r) {
      long long gr = rowBase + wave * 16 + mloc + r;
      if (gr < M) out[(size_t)gr * D + n] = acc[ct][r];
    }
  }
}

// ---------- segment sum + relu (batch is sorted -> binary search, no atomics) ----------
__global__ __launch_bounds__(128) void k_pool(const float* __restrict__ xg,
    const int* __restrict__ batch, float* __restrict__ xm, int N) {
  int g = blockIdx.x, t = threadIdx.x;
  __shared__ int ss[2];
  if (t == 0) { ss[0] = lowerb(batch, N, g); ss[1] = lowerb(batch, N, g + 1); }
  __syncthreads();
  float s = 0.f;
  for (int n = ss[0]; n < ss[1]; ++n) s += xg[(size_t)n * D + t];
  xm[(size_t)g * D + t] = fmaxf(s, 0.f);
}

// ---------- attention: alpha + z = xg@Wred2^T, fused with xg@Wred_l^T + gather ----------
__global__ __launch_bounds__(256) void k_attn(const float* __restrict__ XG,
    const float* __restrict__ Wl, const float* __restrict__ Wz,
    const float* __restrict__ xmr, const int* __restrict__ batch,
    const float* __restrict__ att, float* __restrict__ alpha,
    float* __restrict__ z, int M) {
  __shared__ __bf16 sWl[D * WSTR];
  __shared__ __bf16 sWz[D * WSTR];
  __shared__ __bf16 sX [D * XSTR];
  __shared__ float  sMid[D * MSTR];
  __shared__ float  sAtt[D];
  int tid = threadIdx.x, wave = tid >> 5, lane = tid & 31;
  long long rowBase = (long long)blockIdx.x * 128;
  stage_w(sWl, Wl, tid);
  stage_w(sWz, Wz, tid);
  stage_x(sX, XG, nullptr, rowBase, M, tid);
  if (tid < D) sAtt[tid] = att[tid];
  __syncthreads();

  v8f acc[8];
  gemm_tile(sX, sWl, wave, lane, acc);
  int mloc = (lane >> 4) << 3;
  int bidx[8];
  #pragma unroll
  for (int r = 0; r < 8; ++r) {
    long long gr = rowBase + wave * 16 + mloc + r;
    bidx[r] = (gr < M) ? batch[gr] : 0;
  }
  #pragma unroll
  for (int ct = 0; ct < 8; ++ct) {
    int n = ct * 16 + (lane & 15);
    #pragma unroll
    for (int r = 0; r < 8; ++r) {
      float v = acc[ct][r] + xmr[(size_t)bidx[r] * D + n];
      sMid[(wave * 16 + mloc + r) * MSTR + n] = leakyf(v);
    }
  }
  __syncthreads();

  if (tid < 128) {                                    // alpha = leaky(dot(xji, att))
    long long gr = rowBase + tid;
    float s = 0.f;
    #pragma unroll 8
    for (int j = 0; j < D; ++j) s += sMid[tid * MSTR + j] * sAtt[j];
    if (gr < M) alpha[gr] = leakyf(s);
  }

  gemm_tile(sX, sWz, wave, lane, acc);                // z = xg @ Wred2^T
  #pragma unroll
  for (int ct = 0; ct < 8; ++ct) {
    int n = ct * 16 + (lane & 15);
    #pragma unroll
    for (int r = 0; r < 8; ++r) {
      long long gr = rowBase + wave * 16 + mloc + r;
      if (gr < M) z[(size_t)gr * D + n] = acc[ct][r];
    }
  }
}

// ---------- segment softmax + weighted sum + elu + LayerNorm ----------
__global__ __launch_bounds__(128) void k_soft(const float* __restrict__ z,
    const float* __restrict__ alpha, const int* __restrict__ batch,
    const float* __restrict__ xm, const float* __restrict__ bias,
    float* __restrict__ xmf, int N) {
  int g = blockIdx.x, t = threadIdx.x;
  __shared__ float red[128];
  __shared__ int ss[2];
  if (t == 0) { ss[0] = lowerb(batch, N, g); ss[1] = lowerb(batch, N, g + 1); }
  __syncthreads();
  int s0 = ss[0], s1 = ss[1];

  float mx = -3.4e38f;
  for (int n = s0 + t; n < s1; n += 128) mx = fmaxf(mx, alpha[n]);
  red[t] = mx; __syncthreads();
  for (int o = 64; o; o >>= 1) { if (t < o) red[t] = fmaxf(red[t], red[t + o]); __syncthreads(); }
  mx = red[0]; __syncthreads();

  float sm = 0.f;
  for (int n = s0 + t; n < s1; n += 128) sm += __expf(alpha[n] - mx);
  red[t] = sm; __syncthreads();
  for (int o = 64; o; o >>= 1) { if (t < o) red[t] += red[t + o]; __syncthreads(); }
  sm = red[0] + 1e-16f; __syncthreads();

  float acc = 0.f;
  for (int n = s0; n < s1; ++n) {
    __builtin_prefetch(&z[(size_t)(n + 4) * D + t], 0, 1);
    acc += z[(size_t)n * D + t] * (__expf(alpha[n] - mx) / sm);
  }
  float v = xm[(size_t)g * D + t] + acc + bias[t];
  v = eluf(v);

  red[t] = v; __syncthreads();
  for (int o = 64; o; o >>= 1) { if (t < o) red[t] += red[t + o]; __syncthreads(); }
  float mean = red[0] * (1.f / D); __syncthreads();
  float d = v - mean;
  red[t] = d * d; __syncthreads();
  for (int o = 64; o; o >>= 1) { if (t < o) red[t] += red[t + o]; __syncthreads(); }
  float var = red[0] * (1.f / D);
  xmf[(size_t)g * D + t] = d * __frsqrt_rn(var + 1e-5f);
}

// ---------- host ----------
extern "C" void kernel_launch(void* const* d_in, const int* in_sizes, int n_in,
                              void* d_out, int out_size, void* d_ws, size_t ws_size,
                              hipStream_t stream) {
  const float* x_centers = (const float*)d_in[0];
  const float* x_global  = (const float*)d_in[1];
  const int*   batch     = (const int*)d_in[2];
  const float* Wt_loc1   = (const float*)d_in[3];
  const float* Wt_loc2   = (const float*)d_in[4];
  const float* Wt_glob1  = (const float*)d_in[5];
  const float* Wt_glob2  = (const float*)d_in[6];
  const float* Wred_l    = (const float*)d_in[7];
  const float* Wred_r    = (const float*)d_in[8];
  const float* att       = (const float*)d_in[9];
  const float* Wred2     = (const float*)d_in[10];
  const float* bias_red  = (const float*)d_in[11];
  const float* Wexit     = (const float*)d_in[12];
  const int G = in_sizes[0] / D;
  const int N = in_sizes[1] / D;

  char* ws = (char*)d_ws; size_t off = 0;
  auto alloc = [&](size_t bytes) -> float* {
    float* p = (float*)(ws + off);
    off += (bytes + 255) & ~(size_t)255;
    return p;
  };
  float* xc  = alloc((size_t)G * D * 4);
  float* xg  = alloc((size_t)N * D * 4);
  float* xm  = alloc((size_t)G * D * 4);
  float* xmr = alloc((size_t)G * D * 4);
  float* alp = alloc((size_t)N * 4);
  float* z   = alloc((size_t)N * D * 4);
  float* xmf = alloc((size_t)G * D * 4);
  (void)ws_size; (void)n_in; (void)out_size;

  const int gB = (G + 127) / 128;
  const int nB = (N + 127) / 128;

  k_transfer<<<gB, 256, 0, stream>>>(x_centers, Wt_loc1, Wt_loc2, xc, G);
  k_transfer<<<nB, 256, 0, stream>>>(x_global, Wt_glob1, Wt_glob2, xg, N);
  k_pool    <<<G, 128, 0, stream>>>(xg, batch, xm, N);
  k_gemm    <<<gB, 256, 0, stream>>>(xm, nullptr, Wred_r, xmr, G);
  k_attn    <<<nB, 256, 0, stream>>>(xg, Wred_l, Wred2, xmr, batch, att, alp, z, N);
  k_soft    <<<G, 128, 0, stream>>>(z, alp, batch, xm, bias_red, xmf, N);
  k_gemm    <<<gB, 256, 0, stream>>>(xc, xmf, Wexit, (float*)d_out, G);
}
